// SAUTE_56384330662225
// MI455X (gfx1250) — matrix operand order repaired and checked by reading, not compiled, not run on hardware
//
#include <hip/hip_runtime.h>
#include <stdint.h>

// ---------------- problem constants ----------------
#define B_ 8
#define T_ 256
#define L_ 64
#define D_ 768
#define S_ 8
#define ROWS 32                 // output-column slice per workgroup
#define SLICES (D_ / ROWS)      // 24
#define J2 (D_ / 2)             // 384 packed-bf16-pair words per row

#define USE_TDM 1               // tensor_load_to_lds for the bulk A-tile copy

typedef __attribute__((ext_vector_type(16))) __bf16 v16bf;
typedef __attribute__((ext_vector_type(8)))  float  v8f;
typedef uint32_t u32x4 __attribute__((ext_vector_type(4)));
typedef uint32_t u32x8 __attribute__((ext_vector_type(8)));

union ABFrag { uint32_t u[8]; v16bf v; };

// fast f32x2 -> packed bf16 pair: round-half-up + single v_perm_b32
__device__ __forceinline__ uint32_t pkbf(float lo, float hi) {
  uint32_t a = __float_as_uint(lo) + 0x8000u;
  uint32_t b = __float_as_uint(hi) + 0x8000u;
  return __builtin_amdgcn_perm(b, a, 0x07060302u);  // {hi[31:16], lo[31:16]}
}

__device__ __forceinline__ void wait_asynccnt0() {
#if __has_builtin(__builtin_amdgcn_s_wait_asynccnt)
  __builtin_amdgcn_s_wait_asynccnt(0);
#else
  asm volatile("s_wait_asynccnt 0x0" ::: "memory");
#endif
}

__device__ __forceinline__ void wait_tensorcnt0() {
#if __has_builtin(__builtin_amdgcn_s_wait_tensorcnt)
  __builtin_amdgcn_s_wait_tensorcnt(0);
#else
  asm volatile("s_wait_tensorcnt 0x0" ::: "memory");
#endif
}

// per-lane async copy of 16B from global to LDS (ASYNCcnt-tracked)
__device__ __forceinline__ void async_b128(uint32_t ldsa, const void* gaddr) {
  asm volatile("global_load_async_to_lds_b128 %0, %1, off"
               :: "v"(ldsa), "v"(gaddr) : "memory");
}

// TDM: DMA a contiguous run of `ndw` dwords (ndw < 65536) from global to LDS.
// D# built per CDNA5 ISA 8.3/8.4: 1-row tile, data_size=4B, no cluster mask.
__device__ __forceinline__ void tdm_load_1d(uint32_t lds_addr,
                                            const void* gptr, uint32_t ndw) {
  uint64_t ga = (uint64_t)(uintptr_t)gptr;
  u32x4 g0;
  g0[0] = 1u;                                               // count=1, user D#
  g0[1] = lds_addr;                                         // lds_addr[31:0]
  g0[2] = (uint32_t)ga;                                     // global_addr lo
  g0[3] = ((uint32_t)(ga >> 32) & 0x01FFFFFFu) | 0x80000000u;  // addr hi|type=2
  u32x8 g1;
  g1[0] = 2u << 16;                   // workgroup_mask=0, data_size=4B
  g1[1] = (ndw & 0xFFFFu) << 16;      // tensor_dim0 lo16  (abar addr = 0)
  g1[2] = ((ndw >> 16) & 0xFFFFu) | (1u << 16);  // tensor_dim0 hi, tensor_dim1=1
  g1[3] = (ndw & 0xFFFFu) << 16;      // tile_dim0 (16-bit)
  g1[4] = 1u;                         // tile_dim1=1, tile_dim2 unused
  g1[5] = ndw;                        // tensor_dim0_stride lo32
  g1[6] = (ndw & 0xFFFFu) << 16;      // stride0 hi16=0, stride1 lo16
  g1[7] = 0u;                         // stride1 hi32
  asm volatile("tensor_load_to_lds %0, %1" :: "s"(g0), "s"(g1) : "memory");
}

// ---------------------------------------------------------------------------
// Kernel 0: one-time f32 -> packed-bf16 conversion of token_embeddings.
// ---------------------------------------------------------------------------
__global__ __launch_bounds__(256) void tok_bf16_kernel(
    const float4* __restrict__ in, uint4* __restrict__ outw) {
  int i = blockIdx.x * 256 + threadIdx.x;   // one uint4 (8 floats) per thread
  float4 a = in[2 * i + 0];
  float4 b = in[2 * i + 1];
  uint4  r;
  r.x = pkbf(a.x, a.y);
  r.y = pkbf(a.z, a.w);
  r.z = pkbf(b.x, b.y);
  r.w = pkbf(b.z, b.w);
  outw[i] = r;
}

// ---------------------------------------------------------------------------
// Kernel 1: k = edu @ Wk^T, v = edu @ Wv^T   (fp32, tiled; ~3% of FLOPs)
// ---------------------------------------------------------------------------
__global__ __launch_bounds__(256) void saute_proj_kernel(
    const float* __restrict__ edu, const float* __restrict__ Wk,
    const float* __restrict__ Wv, float* __restrict__ kout,
    float* __restrict__ vout) {
  __shared__ float es[64][65];
  __shared__ float ws_k[64][65];
  __shared__ float ws_v[64][65];
  const int tid = threadIdx.x;
  const int bt0 = blockIdx.x * 64;
  const int e0  = blockIdx.y * 64;
  const int tx = tid & 15, ty = tid >> 4;

  float ak[4][4] = {{0.f}}, av[4][4] = {{0.f}};

  for (int kc = 0; kc < D_; kc += 64) {
    __syncthreads();
#pragma unroll
    for (int n = 0; n < 16; ++n) {
      int idx = tid + 256 * n;
      int r = idx >> 6, c = idx & 63;
      es[r][c]   = edu[(size_t)(bt0 + r) * D_ + kc + c];
      ws_k[r][c] = Wk[(size_t)(e0 + r) * D_ + kc + c];
      ws_v[r][c] = Wv[(size_t)(e0 + r) * D_ + kc + c];
    }
    __syncthreads();
#pragma unroll 8
    for (int dd = 0; dd < 64; ++dd) {
      float a[4], bk[4], bv[4];
#pragma unroll
      for (int i = 0; i < 4; ++i) a[i] = es[ty * 4 + i][dd];
#pragma unroll
      for (int j = 0; j < 4; ++j) { bk[j] = ws_k[tx * 4 + j][dd]; bv[j] = ws_v[tx * 4 + j][dd]; }
#pragma unroll
      for (int i = 0; i < 4; ++i)
#pragma unroll
        for (int j = 0; j < 4; ++j) { ak[i][j] += a[i] * bk[j]; av[i][j] += a[i] * bv[j]; }
    }
  }
#pragma unroll
  for (int i = 0; i < 4; ++i)
#pragma unroll
    for (int j = 0; j < 4; ++j) {
      size_t off = (size_t)(bt0 + ty * 4 + i) * D_ + (e0 + tx * 4 + j);
      kout[off] = ak[i][j];
      vout[off] = av[i][j];
    }
}

// ---------------------------------------------------------------------------
// Kernel 2: sequential speaker-state scan. WG = (batch, speaker, 32-col slice)
// owns its rows of M[s] exclusively -> fp32 state lives wholly in LDS (CDNA5
// 320KB WGP LDS), no inter-WG sync. A-tile via TDM, k/v via async-to-LDS,
// GEMM via v_wmma_f32_16x16x32_bf16.
// ---------------------------------------------------------------------------
__global__ __launch_bounds__(256, 1) void saute_scan_kernel(
    const int* __restrict__ spk_ids,      // [B,T]
    const float* __restrict__ tok,        // [B,T,L,D] fp32
    const uint32_t* __restrict__ tokbf,   // [B,T,L,J2] bf16-pairs (may be null)
    const float* __restrict__ kproj,      // [B,T,D]
    const float* __restrict__ vproj,      // [B,T,D]
    float* __restrict__ out,              // [B,T,L,D]
    int preconv) {
  extern __shared__ char smem[];
  float*    state = (float*)smem;                  // [ROWS][D]   fp32   98304 B
  uint32_t* Bb    = (uint32_t*)(smem + 98304);     // [ROWS][J2]  bf16x2 49152 B
  uint32_t* Ab    = (uint32_t*)(smem + 147456);    // [L][J2]     bf16x2 98304 B
  float*    vbuf  = (float*)(smem + 245760);       // [D]                 3072 B
  float*    kbuf  = (float*)(smem + 248832);       // [ROWS]               128 B
  int*      spkl  = (int*)(smem + 248960);         // [T]                 1024 B

  const int tid   = threadIdx.x;
  const int slice = blockIdx.x % SLICES;
  const int s     = (blockIdx.x / SLICES) % S_;
  const int b     = blockIdx.x / (SLICES * S_);
  const int ibase = slice * ROWS;

  spkl[tid] = spk_ids[b * T_ + tid];
  {
    float4  z   = make_float4(0.f, 0.f, 0.f, 0.f);
    float4* st4 = (float4*)state;
#pragma unroll
    for (int n = 0; n < (ROWS * D_ / 4) / 256; ++n) st4[tid + 256 * n] = z;
  }
  __syncthreads();

  const int wave = tid >> 5, lane = tid & 31;
  const int mt = wave & 3, nt = wave >> 2;       // 4 l-tiles x 2 n-tiles
  const int lcol = lane & 15, khalf = lane >> 4;

  for (int t = 0; t < T_; ++t) {
    if (spkl[t] != s) continue;                  // uniform across block
    const int    bt      = b * T_ + t;
    const size_t tokbase = (size_t)bt * (size_t)(L_ * D_);

    __syncthreads();  // prior step's LDS reads complete before restaging

    // ---- stage A tile (tok as packed bf16) ----
    if (preconv) {
#if USE_TDM
      if (wave == 0) {   // one TDM descriptor moves the whole 96KB tile
        tdm_load_1d((uint32_t)(uintptr_t)Ab,
                    tokbf + (size_t)bt * (L_ * J2), L_ * J2);
      }
#else
#pragma unroll
      for (int n = 0; n < (L_ * J2 / 4) / 256; ++n) {   // 24 x b128 per thread
        int c = tid + 256 * n;
        async_b128((uint32_t)(uintptr_t)(const void*)(Ab + c * 4),
                   tokbf + (size_t)bt * (L_ * J2) + c * 4);
      }
#endif
    } else {
      const float4* tp4 = (const float4*)(tok + tokbase);
#pragma unroll
      for (int n = 0; n < (L_ * D_ / 4) / 256; ++n) {   // 48 float4 per thread
        int    c = tid + 256 * n;
        float4 x = tp4[c];
        Ab[c * 2 + 0] = pkbf(x.x, x.y);
        Ab[c * 2 + 1] = pkbf(x.z, x.w);
      }
    }

    // ---- stage k-slice and full v (fp32, async global->LDS) ----
    if (tid < D_ / 4)   // 192 lanes x 16B = 768 floats
      async_b128((uint32_t)(uintptr_t)(vbuf + tid * 4),
                 vproj + (size_t)bt * D_ + tid * 4);
    if (tid < ROWS / 4) // 8 lanes x 16B = 32 floats
      async_b128((uint32_t)(uintptr_t)(kbuf + tid * 4),
                 kproj + (size_t)bt * D_ + ibase + tid * 4);

    // ---- prefetch next matching step's tile (global_prefetch_b8) ----
    {
      int tn = t + 1;
      while (tn < T_ && spkl[tn] != s) ++tn;     // uniform scan, ~S steps
      if (tn < T_) {
        int btn = b * T_ + tn;
        if (preconv) {
          const char* p = (const char*)(tokbf + (size_t)btn * (L_ * J2));
#pragma unroll
          for (int j = 0; j < 3; ++j)            // 768 cachelines / 256 thr
            __builtin_prefetch(p + (tid + 256 * j) * 128, 0, 3);
        } else {
          const char* p = (const char*)(tok + (size_t)btn * (L_ * D_));
#pragma unroll
          for (int j = 0; j < 6; ++j)            // 1536 cachelines / 256 thr
            __builtin_prefetch(p + (tid + 256 * j) * 128, 0, 3);
        }
      }
    }

    wait_asynccnt0();                            // my async LDS writes landed
#if USE_TDM
    if (preconv) wait_tensorcnt0();              // TDM tile landed (wave 0)
#endif
    __syncthreads();

    // ---- rank-1 fp32 update of state rows, fused bf16 repack of B=M^T ----
#pragma unroll
    for (int n = 0; n < (ROWS * J2) / 256; ++n) {   // 48 pairs per thread
      int   p  = tid + 256 * n;
      int   i  = p / J2, j2 = p - i * J2;
      float kk = kbuf[i];
      float s0 = state[i * D_ + 2 * j2 + 0] + kk * vbuf[2 * j2 + 0];
      float s1 = state[i * D_ + 2 * j2 + 1] + kk * vbuf[2 * j2 + 1];
      state[i * D_ + 2 * j2 + 0] = s0;
      state[i * D_ + 2 * j2 + 1] = s1;
      Bb[p] = pkbf(s0, s1);
    }
    __syncthreads();

    // ---- out[l, i] = tok[l, i] + sum_j tok[l,j] * M[i,j] via WMMA ----
    v8f acc = {0.f, 0.f, 0.f, 0.f, 0.f, 0.f, 0.f, 0.f};
    const int arow = mt * 16 + lcol;
    const int bcol = nt * 16 + lcol;
#pragma unroll 4
    for (int kb = 0; kb < D_; kb += 32) {
      int    aw = arow * J2 + (kb >> 1) + khalf * 4;
      int    bw = bcol * J2 + (kb >> 1) + khalf * 4;
      ABFrag fa, fb;
      *(uint4*)&fa.u[0] = *(const uint4*)(Ab + aw);
      *(uint4*)&fa.u[4] = *(const uint4*)(Ab + aw + 8);
      *(uint4*)&fb.u[0] = *(const uint4*)(Bb + bw);
      *(uint4*)&fb.u[4] = *(const uint4*)(Bb + bw + 8);
      acc = __builtin_amdgcn_wmma_f32_16x16x32_bf16(false, fa.v, false, fb.v,
                                                    (short)0, acc, false, false);
    }
    {
      const int    col   = ibase + nt * 16 + lcol;
      const int    rbase = mt * 16 + khalf * 8;
      const float* tr    = tok + tokbase;
      float*       orow  = out + tokbase;
#pragma unroll
      for (int r = 0; r < 8; ++r) {
        size_t off = (size_t)(rbase + r) * D_ + col;
        orow[off]  = tr[off] + acc[r];           // fp32 residual from global
      }
    }
  }
}

// ---------------------------------------------------------------------------
extern "C" void kernel_launch(void* const* d_in, const int* in_sizes, int n_in,
                              void* d_out, int out_size, void* d_ws,
                              size_t ws_size, hipStream_t stream) {
  (void)in_sizes; (void)n_in; (void)out_size;
  const int*   spk = (const int*)d_in[1];
  const float* tok = (const float*)d_in[2];
  const float* edu = (const float*)d_in[3];
  const float* Wk  = (const float*)d_in[4];
  const float* Wv  = (const float*)d_in[5];
  float*       out = (float*)d_out;

  const size_t kvBytes    = 2ull * B_ * T_ * D_ * sizeof(float);       // 12.6 MB
  const size_t tokbfWords = (size_t)B_ * T_ * L_ * J2;                 // 50.3 M
  const size_t needBytes  = kvBytes + tokbfWords * sizeof(uint32_t);   // ~214 MB

  float*    kws   = (float*)d_ws;
  float*    vws   = kws + (size_t)B_ * T_ * D_;
  uint32_t* tokbf = (uint32_t*)((char*)d_ws + kvBytes);
  const int preconv = (ws_size >= needBytes) ? 1 : 0;

  const size_t smem2 = 249984;   // 244 KB dynamic LDS (CDNA5 WGP has 320 KB)
  (void)hipFuncSetAttribute((const void*)saute_scan_kernel,
                            hipFuncAttributeMaxDynamicSharedMemorySize,
                            (int)smem2);

  if (preconv) {
    int n4 = (int)(tokbfWords / 4);              // uint4 chunks
    tok_bf16_kernel<<<dim3(n4 / 256), dim3(256), 0, stream>>>(
        (const float4*)tok, (uint4*)tokbf);
  }

  dim3 g1((B_ * T_) / 64, D_ / 64);
  saute_proj_kernel<<<g1, dim3(256), 0, stream>>>(edu, Wk, Wv, kws, vws);

  saute_scan_kernel<<<dim3(B_ * S_ * SLICES), dim3(256), smem2, stream>>>(
      spk, tok, tokbf, kws, vws, out, preconv);
}
